// TransformerEncoder_24661702214069
// MI455X (gfx1250) — matrix operand order repaired and checked by reading, not compiled, not run on hardware
//
#include <hip/hip_runtime.h>
#include <cstdint>

// ---------------------------------------------------------------------------
// Types for CDNA5 WMMA (wave32): v_wmma_f32_16x16x32_f16
// ---------------------------------------------------------------------------
typedef _Float16 v16h  __attribute__((ext_vector_type(16)));
typedef _Float16 half8 __attribute__((ext_vector_type(8)));
typedef _Float16 hv4   __attribute__((ext_vector_type(4)));
typedef float    v8f   __attribute__((ext_vector_type(8)));
typedef float    fv4   __attribute__((ext_vector_type(4)));

#define DEVINL __device__ __forceinline__

constexpr int Lq = 2048, Nb = 2, E = 512, Hh = 8, HD = 64, FF = 2048, NL = 2;
constexpr int Mtok = Lq * Nb;  // 4096 tokens, row index m = l*Nb + n

// Async global->LDS 16B copy (GLOBAL_LOAD_ASYNC_TO_LDS_B128, ASYNCcnt).
// lds_off = byte offset into workgroup LDS (generic-pointer low 32 bits).
DEVINL void async_copy_b128(unsigned lds_off, const _Float16* g) {
  asm volatile("global_load_async_to_lds_b128 %0, %1, off" ::"v"(lds_off),
               "v"(g)
               : "memory");
}
DEVINL void wait_async_le1() {
  asm volatile("s_wait_asynccnt 0x1" ::: "memory");
}
DEVINL void wait_async_le0() {
  asm volatile("s_wait_asynccnt 0x0" ::: "memory");
}

// A-matrix fragment: 16x32 (MxK) f16, row-major source with leading dim lda.
// lanes 0-15 -> M=lane, halves 0..15 = K 0..15
// lanes 16-31 -> M=lane-16, halves 0..15 = K 16..31
// Lowers to two global_load_b128 into adjacent VGPRs.
DEVINL v16h load_a_frag(const _Float16* __restrict__ A, int lda, int m0, int k0,
                        int lane) {
  int m  = m0 + (lane & 15);
  int kb = (lane >> 4) << 4;  // 0 or 16
  const _Float16* p = A + (size_t)m * lda + (k0 + kb);
  half8 lo = *(const half8*)(p);
  half8 hi = *(const half8*)(p + 8);
  v16h r;
#pragma unroll
  for (int i = 0; i < 8; ++i) { r[i] = lo[i]; r[i + 8] = hi[i]; }
  return r;
}

// B fragment from the LDS-staged tile shB[64][32] (W^T tile, rows = out col).
// lanes 0-15: K=0..15, lanes 16-31: K=16..31; col = lane&15 -> ds_load_b128 x2.
DEVINL v16h load_b_lds(const _Float16* tile, int f, int lane) {
  return *(const v16h*)(tile + (size_t)f * 32 + ((lane >> 4) << 4));
}

// ---------------------------------------------------------------------------
// Generic batched WMMA GEMM:  D = act(scale * A @ W^T + bias)
//   A: f16 [Mr, K] (lda); W: f16 [Ncols, K] row-major; D: f16 or f32 (ldc)
// Block: 256 threads = 8 waves, block tile 256(M) x 64(N); wave tile 64x32
// = 4x2 accumulators -> 8 v_wmma per K-step. The 64x32 B tile is staged in
// LDS via double-buffered async copies (1 x b128 per thread per K-tile) and
// shared by all 8 waves. All dims tile exactly; all branches wave-uniform.
// ---------------------------------------------------------------------------
template <bool OUTF16>
__global__ __launch_bounds__(256) void gemm_wmma_f16(
    const _Float16* __restrict__ A, const _Float16* B,
    const float* __restrict__ bias, void* __restrict__ C, int K, int lda,
    int ldb, int ldc, long long offA_n, long long offA_h, long long offB_n,
    long long offB_h, long long offC_n, long long offC_h, int HB, float scale,
    int relu) {
  __shared__ _Float16 shB[2][64][32];  // 8 KB double buffer

  const int z  = blockIdx.z;
  const int bh = z % HB;
  const int bn = z / HB;
  const _Float16* Ab = A + (size_t)bn * offA_n + (size_t)bh * offA_h;
  const _Float16* Bb = B + (size_t)bn * offB_n + (size_t)bh * offB_h;
  const size_t cbase = (size_t)bn * offC_n + (size_t)bh * offC_h;

  const int tid  = threadIdx.x;
  const int lane = tid & 31;
  const int wid  = tid >> 5;
  const int m0 = blockIdx.y * 256 + (wid >> 1) * 64;
  const int nw = (wid & 1) * 32;             // wave's n offset inside block tile
  const int n0 = blockIdx.x * 64 + nw;

  // Per-thread async-copy slot: row f (0..63), 16B chunk c (0,8,16,24 halves)
  const int cf = tid >> 2;
  const int cc = (tid & 3) * 8;
  const _Float16* gB = Bb + (size_t)(blockIdx.x * 64 + cf) * ldb + cc;
  const unsigned lds0 = (unsigned)(uintptr_t)&shB[0][cf][cc];
  const unsigned lds1 = (unsigned)(uintptr_t)&shB[1][cf][cc];

  const _Float16* pfa = Ab + (size_t)(m0 + (lane & 15)) * lda;

  v8f acc[4][2] = {};

  async_copy_b128(lds0, gB);  // stage K-tile 0
  int buf = 0;
  for (int k = 0; k < K; k += 32) {
    const bool more = (k + 32) < K;
    if (more) async_copy_b128(buf ? lds0 : lds1, gB + (k + 32));
    if (k + 128 < K) __builtin_prefetch((const void*)(pfa + k + 128), 0, 1);
    if (more) wait_async_le1();  // in-order ASYNCcnt: current tile landed
    else      wait_async_le0();
    __syncthreads();  // tile visible to all 8 waves

    const _Float16* tile = &shB[buf][0][0];
    v16h b0 = load_b_lds(tile, nw + (lane & 15), lane);
    v16h b1 = load_b_lds(tile, nw + 16 + (lane & 15), lane);
#pragma unroll
    for (int i = 0; i < 4; ++i) {
      v16h a = load_a_frag(Ab, lda, m0 + 16 * i, k, lane);
      acc[i][0] = __builtin_amdgcn_wmma_f32_16x16x32_f16(
          false, a, false, b0, (short)0, acc[i][0], false, false);
      acc[i][1] = __builtin_amdgcn_wmma_f32_16x16x32_f16(
          false, a, false, b1, (short)0, acc[i][1], false, false);
    }
    __syncthreads();  // everyone done reading buf before it is re-filled
    buf ^= 1;
  }

  // C/D layout: VGPR r -> row = r + 8*(lane>=16), col = lane&15
  const int cg = lane & 15;
  const int ro = (lane >> 4) << 3;
#pragma unroll
  for (int tn = 0; tn < 2; ++tn) {
    const int col = n0 + tn * 16 + cg;
    const float bv = bias[col];
#pragma unroll
    for (int tm = 0; tm < 4; ++tm) {
#pragma unroll
      for (int r = 0; r < 8; ++r) {
        int row = m0 + tm * 16 + ro + r;
        float v = acc[tm][tn][r] * scale + bv;
        if (relu) v = fmaxf(v, 0.0f);
        size_t idx = cbase + (size_t)row * ldc + col;
        if (OUTF16) ((_Float16*)C)[idx] = (_Float16)v;
        else        ((float*)C)[idx] = v;
      }
    }
  }
}

// ---------------------------------------------------------------------------
// In-place row softmax over f16 logits, cols = 2048. One wave32 per row,
// 8 x half8 (16B) per lane held in registers; single global read + write.
// ---------------------------------------------------------------------------
__global__ __launch_bounds__(256) void softmax_rows_f16(
    _Float16* __restrict__ buf) {
  const int row  = blockIdx.x * (blockDim.x >> 5) + (threadIdx.x >> 5);
  const int lane = threadIdx.x & 31;
  half8* p8 = (half8*)(buf + (size_t)row * Lq);

  half8 v[8];
  float mx = -3.0e38f;
#pragma unroll
  for (int i = 0; i < 8; ++i) {
    v[i] = p8[i * 32 + lane];
#pragma unroll
    for (int e = 0; e < 8; ++e) mx = fmaxf(mx, (float)v[i][e]);
  }
#pragma unroll
  for (int o = 16; o; o >>= 1) mx = fmaxf(mx, __shfl_xor(mx, o, 32));

  float sum = 0.0f;
#pragma unroll
  for (int i = 0; i < 8; ++i) {
#pragma unroll
    for (int e = 0; e < 8; ++e) {
      float ex = __expf((float)v[i][e] - mx);
      sum += ex;
      v[i][e] = (_Float16)ex;
    }
  }
#pragma unroll
  for (int o = 16; o; o >>= 1) sum += __shfl_xor(sum, o, 32);

  float inv = 1.0f / sum;
#pragma unroll
  for (int i = 0; i < 8; ++i) {
#pragma unroll
    for (int e = 0; e < 8; ++e) v[i][e] = (_Float16)((float)v[i][e] * inv);
    p8[i * 32 + lane] = v[i];
  }
}

// ---------------------------------------------------------------------------
// y = LayerNorm(x + r) * g + b ; writes f32 (next residual) and f16 (GEMM in).
// One wave32 per row of 512; float4 vector accesses, 4 per lane.
// ---------------------------------------------------------------------------
__global__ __launch_bounds__(256) void residual_layernorm(
    const float* __restrict__ x, const float* __restrict__ r,
    const float* __restrict__ g, const float* __restrict__ bb,
    float* __restrict__ y, _Float16* __restrict__ y16) {
  const int row  = blockIdx.x * (blockDim.x >> 5) + (threadIdx.x >> 5);
  const int lane = threadIdx.x & 31;
  const fv4* px = (const fv4*)(x + (size_t)row * E);
  const fv4* pr = (const fv4*)(r + (size_t)row * E);
  const fv4* pg = (const fv4*)g;
  const fv4* pb = (const fv4*)bb;
  fv4* py   = (fv4*)(y + (size_t)row * E);
  hv4* py16 = (hv4*)(y16 + (size_t)row * E);

  fv4 v[4];
  float s = 0.0f, sq = 0.0f;
#pragma unroll
  for (int i = 0; i < 4; ++i) {
    int c = i * 32 + lane;
    fv4 t = px[c] + pr[c];
    v[i] = t;
#pragma unroll
    for (int e = 0; e < 4; ++e) { s += t[e]; sq += t[e] * t[e]; }
  }
#pragma unroll
  for (int o = 16; o; o >>= 1) {
    s  += __shfl_xor(s, o, 32);
    sq += __shfl_xor(sq, o, 32);
  }
  float mu  = s * (1.0f / E);
  float var = sq * (1.0f / E) - mu * mu;
  float inv = rsqrtf(var + 1e-5f);
#pragma unroll
  for (int i = 0; i < 4; ++i) {
    int c = i * 32 + lane;
    fv4 gg = pg[c], bbv = pb[c];
    fv4 t;
    hv4 th;
#pragma unroll
    for (int e = 0; e < 4; ++e) {
      t[e]  = (v[i][e] - mu) * inv * gg[e] + bbv[e];
      th[e] = (_Float16)t[e];
    }
    py[c]   = t;
    py16[c] = th;
  }
}

// 8 elements per thread: two float4 loads -> one half8 store.
__global__ void cvt_f32_to_f16(const float* __restrict__ in,
                               _Float16* __restrict__ out, long long n8) {
  long long i = (long long)blockIdx.x * blockDim.x + threadIdx.x;
  if (i >= n8) return;
  const fv4* p = (const fv4*)in + i * 2;
  fv4 a = p[0], b = p[1];
  half8 h;
#pragma unroll
  for (int e = 0; e < 4; ++e) { h[e] = (_Float16)a[e]; h[e + 4] = (_Float16)b[e]; }
  ((half8*)out)[i] = h;
}

// vt[n][h][d][s] = V[s][n][h][d]  (V lives in qkv16 at column offset 2E)
__global__ void transpose_v(const _Float16* __restrict__ qkv,
                            _Float16* __restrict__ vt) {
  int idx = blockIdx.x * blockDim.x + threadIdx.x;  // over Nb*Hh*HD*Lq = 2M
  int s = idx & (Lq - 1);
  int d = (idx >> 11) & (HD - 1);
  int h = (idx >> 17) & (Hh - 1);
  int n = idx >> 20;
  vt[idx] = qkv[(size_t)(s * Nb + n) * (3 * E) + 2 * E + h * HD + d];
}

__global__ void zero_f32(float* __restrict__ p, int n) {
  int i = blockIdx.x * blockDim.x + threadIdx.x;
  if (i < n) p[i] = 0.0f;
}

// ---------------------------------------------------------------------------
// Host orchestration
// ---------------------------------------------------------------------------
extern "C" void kernel_launch(void* const* d_in, const int* in_sizes, int n_in,
                              void* d_out, int out_size, void* d_ws,
                              size_t ws_size, hipStream_t stream) {
  (void)in_sizes; (void)n_in; (void)out_size; (void)ws_size;

  const float* src  = (const float*)d_in[0];
  const float* wqkv = (const float*)d_in[1];
  const float* bqkv = (const float*)d_in[2];
  const float* wout = (const float*)d_in[3];
  const float* bout = (const float*)d_in[4];
  const float* w1   = (const float*)d_in[5];
  const float* b1   = (const float*)d_in[6];
  const float* w2   = (const float*)d_in[7];
  const float* b2   = (const float*)d_in[8];
  const float* g1   = (const float*)d_in[9];
  const float* be1  = (const float*)d_in[10];
  const float* g2   = (const float*)d_in[11];
  const float* be2  = (const float*)d_in[12];

  // Workspace carve-up (f16 region first, then f32 region). ~205 MB total.
  _Float16* hp = (_Float16*)d_ws;
  _Float16* qkvw16 = hp; hp += (size_t)NL * 3 * E * E;
  _Float16* outw16 = hp; hp += (size_t)NL * E * E;
  _Float16* l1w16  = hp; hp += (size_t)NL * FF * E;
  _Float16* l2w16  = hp; hp += (size_t)NL * E * FF;
  _Float16* xf16   = hp; hp += (size_t)Mtok * E;
  _Float16* qkv16  = hp; hp += (size_t)Mtok * 3 * E;
  _Float16* vt16   = hp; hp += (size_t)Nb * Hh * HD * Lq;
  _Float16* att16  = hp; hp += (size_t)Mtok * E;
  _Float16* h16    = hp; hp += (size_t)Mtok * FF;
  _Float16* prob   = hp; hp += (size_t)Nb * Hh * Lq * Lq;  // [n][h][L][L] f16
  float* xf32  = (float*)hp;
  float* tmpf  = xf32 + (size_t)Mtok * E;
  float* zbias = tmpf + (size_t)Mtok * E;  // FF zeros

  zero_f32<<<dim3(FF / 256), dim3(256), 0, stream>>>(zbias, FF);

  auto cvt = [&](const float* in, _Float16* out, long long n) {
    long long n8 = n / 8;
    cvt_f32_to_f16<<<dim3((unsigned)((n8 + 255) / 256)), dim3(256), 0, stream>>>(
        in, out, n8);
  };
  cvt(wqkv, qkvw16, (long long)NL * 3 * E * E);
  cvt(wout, outw16, (long long)NL * E * E);
  cvt(w1,   l1w16,  (long long)NL * FF * E);
  cvt(w2,   l2w16,  (long long)NL * E * FF);
  cvt(src,  xf16,   (long long)Mtok * E);
  hipMemcpyAsync(xf32, src, (size_t)Mtok * E * sizeof(float),
                 hipMemcpyDeviceToDevice, stream);

  auto gemm = [&](bool outf16, const _Float16* A, const _Float16* B,
                  const float* bias, void* C, int Mr, int Nc, int K, int lda,
                  int ldb, int ldc, long long oAn, long long oAh,
                  long long oBn, long long oBh, long long oCn, long long oCh,
                  int batches, int HB, float scale, int relu) {
    dim3 grid(Nc / 64, Mr / 256, batches);
    dim3 blk(256);
    if (outf16)
      gemm_wmma_f16<true><<<grid, blk, 0, stream>>>(
          A, B, bias, C, K, lda, ldb, ldc, oAn, oAh, oBn, oBh, oCn, oCh, HB,
          scale, relu);
    else
      gemm_wmma_f16<false><<<grid, blk, 0, stream>>>(
          A, B, bias, C, K, lda, ldb, ldc, oAn, oAh, oBn, oBh, oCn, oCh, HB,
          scale, relu);
  };

  for (int i = 0; i < NL; ++i) {
    const _Float16* Wq = qkvw16 + (size_t)i * 3 * E * E;
    const _Float16* Wo = outw16 + (size_t)i * E * E;
    const _Float16* W1 = l1w16 + (size_t)i * FF * E;
    const _Float16* W2 = l2w16 + (size_t)i * E * FF;

    // (1) QKV projection: qkv16[Mtok, 3E] = xf16 @ Wq^T + bq
    gemm(true, xf16, Wq, bqkv + (size_t)i * 3 * E, qkv16, Mtok, 3 * E, E, E, E,
         3 * E, 0, 0, 0, 0, 0, 0, 1, 1, 1.0f, 0);

    // (1b) V transpose: vt[n][h][d][s]
    transpose_v<<<dim3((Nb * Hh * HD * Lq) / 256), dim3(256), 0, stream>>>(
        qkv16, vt16);

    // (2) scores: prob[n][h][L][L] = 0.125 * Q_nh @ K_nh^T  (batched over n,h)
    gemm(true, qkv16, qkv16 + E, zbias, prob, Lq, Lq, HD, Nb * 3 * E,
         Nb * 3 * E, Lq, 3 * E, HD, 3 * E, HD, (long long)Hh * Lq * Lq,
         (long long)Lq * Lq, Nb * Hh, Hh, 0.125f, 0);

    // (3) softmax over each of the N*H*L rows (in place, f16)
    softmax_rows_f16<<<dim3((Nb * Hh * Lq) / 8), dim3(256), 0, stream>>>(prob);

    // (4) attention: att16[(l*Nb+n)*E + h*HD + d] = P_nh @ (Vt_nh)^T
    //     Vt is [HD, Lq] row-major per (n,h) -> contiguous wt loader.
    gemm(true, prob, vt16, zbias, att16, Lq, HD, Lq, Lq, Lq, Nb * E,
         (long long)Hh * Lq * Lq, (long long)Lq * Lq,
         (long long)Hh * HD * Lq, (long long)HD * Lq, E, HD, Nb * Hh, Hh,
         1.0f, 0);

    // (5) out_proj -> tmpf (f32) for the residual
    gemm(false, att16, Wo, bout + (size_t)i * E, tmpf, Mtok, E, E, E, E, E, 0,
         0, 0, 0, 0, 0, 1, 1, 1.0f, 0);

    // (6) x = LN(x + attn_out); writes xf32 + xf16
    residual_layernorm<<<dim3(Mtok / 8), dim3(256), 0, stream>>>(
        xf32, tmpf, g1 + (size_t)i * E, be1 + (size_t)i * E, xf32, xf16);

    // (7) lin1 + ReLU -> h16 (f16)
    gemm(true, xf16, W1, b1 + (size_t)i * FF, h16, Mtok, FF, E, E, E, FF, 0, 0,
         0, 0, 0, 0, 1, 1, 1.0f, 1);

    // (8) lin2 -> tmpf (f32)
    gemm(false, h16, W2, b2 + (size_t)i * E, tmpf, Mtok, E, FF, FF, FF, E, 0,
         0, 0, 0, 0, 0, 1, 1, 1.0f, 0);

    // (9) x = LN(x + ffn_out)
    residual_layernorm<<<dim3(Mtok / 8), dim3(256), 0, stream>>>(
        xf32, tmpf, g2 + (size_t)i * E, be2 + (size_t)i * E, xf32, xf16);
  }

  hipMemcpyAsync(d_out, xf32, (size_t)Mtok * E * sizeof(float),
                 hipMemcpyDeviceToDevice, stream);
}
